// SingleAttnGRU_74122545594411
// MI455X (gfx1250) — matrix operand order repaired and checked by reading, not compiled, not run on hardware
//
#include <hip/hip_runtime.h>
#include <hip/hip_bf16.h>

// ---------------- problem constants ----------------
#define S_LEN 256
#define BATCH 64
#define IN    1024
#define HID   1024
#define H3    3072
#define PDIM  1024
#define PLEN  256
#define NEGV  (-1000000000.0f)

typedef __attribute__((ext_vector_type(16))) _Float16 v16h;
typedef __attribute__((ext_vector_type(8)))  _Float16 v8h;
typedef __attribute__((ext_vector_type(8)))  float    v8f;

// output layout (floats): [h_last: B*H][hs: S*B*2048][attn: S*PLEN*B]
#define HS_BASE   ((size_t)BATCH * HID)
#define ATTN_BASE (HS_BASE + (size_t)S_LEN * BATCH * 2048)

// ---------------- fp32 -> fp16 conversion ----------------
__global__ void cvt_f32_to_f16(const float* __restrict__ src,
                               _Float16* __restrict__ dst, int n) {
    int i = blockIdx.x * blockDim.x + threadIdx.x;
    if (i < n) dst[i] = (_Float16)src[i];
}

// h[b, n] = h_init[n]  (broadcast over batch), both f32 and f16 copies
__global__ void init_h(const float* __restrict__ h_init, float* __restrict__ h,
                       _Float16* __restrict__ h16) {
    int idx = blockIdx.x * blockDim.x + threadIdx.x;   // 0 .. B*H-1
    float v = h_init[idx & (HID - 1)];
    h[idx] = v;
    h16[idx] = (_Float16)v;
}

// ---------------- WMMA GEMM:  C[64,N] = A[64,K](f16) x W[N,K](f16)^T + bias ----
// One wave computes a 16(M) x 128(N) strip: 8 f32 accumulators, K in steps of 32.
// A fragment (16-bit A 16x32, this code's contiguous k order): lane = hi*16 + m;
//   halves 0..7  <- A[m, k0 + 16*hi .. +7]
//   halves 8..15 <- A[m, k0 + 16*hi + 8 .. +7]
// B fragment (32x16): lane&15 = column, 16 contiguous K halves starting k0+16*hi.
__global__ __launch_bounds__(128)
void gemm_wmma_f16(const _Float16* __restrict__ A, int lda,
                   const _Float16* __restrict__ W,      // [N,K] row-major f16
                   const float* __restrict__ bias,
                   float* __restrict__ C, int N, int K) {
    const int wave  = threadIdx.x >> 5;
    const int lane  = threadIdx.x & 31;
    const int hi    = lane >> 4;        // K-half selector for A/B fragments
    const int l15   = lane & 15;        // A: row-in-tile ; B/C: col-in-tile
    const int strip = blockIdx.x * 4 + wave;
    const int mt    = strip & 3;        // 4 M-tiles cover B=64
    const int nt    = strip >> 2;       // N/128 strips
    const int m0    = mt << 4;
    const int n0    = nt << 7;
    if (n0 >= N) return;

    const _Float16* arow = A + (size_t)(m0 + l15) * lda + hi * 16;
    const _Float16* wcol = W + (size_t)(n0 + l15) * K + hi * 16;
    const size_t    wstep = (size_t)16 * K;       // advance 16 N-columns

    v8f acc[8] = {};

    for (int k0 = 0; k0 < K; k0 += 32) {
        v8h a_lo = *(const v8h*)(arow + k0);
        v8h a_hi = *(const v8h*)(arow + k0 + 8);
        v16h a;
#pragma unroll
        for (int j = 0; j < 8; ++j) { a[j] = a_lo[j]; a[j + 8] = a_hi[j]; }
        __builtin_prefetch(arow + k0 + 32, 0, 3);   // global_prefetch_b8 (near)
        __builtin_prefetch(wcol + k0 + 32, 0, 3);
#pragma unroll
        for (int s = 0; s < 8; ++s) {
            v16h b = *(const v16h*)(wcol + (size_t)s * wstep + k0);
            acc[s] = __builtin_amdgcn_wmma_f32_16x16x32_f16(
                         false, a, false, b, (short)0, acc[s], false, false);
        }
    }

    // C/D layout: VGPR r, lane: N = col, M = m0 + r + 8*hi
#pragma unroll
    for (int s = 0; s < 8; ++s) {
        const int ncol = n0 + s * 16 + l15;
        const float bs = bias[ncol];
#pragma unroll
        for (int r = 0; r < 8; ++r) {
            C[(size_t)(m0 + r + 8 * hi) * N + ncol] = acc[s][r] + bs;
        }
    }
}

// ---------------- GRU gate fusion ----------------
// h_new = (n + z*(h-n)) * keep ; updates h (f32+f16) in place, writes hs[:, :, 0:H]
__global__ void gru_gates(const float* __restrict__ gi, const float* __restrict__ gh,
                          float* __restrict__ h, _Float16* __restrict__ h16,
                          const int* __restrict__ length,
                          int t, float* __restrict__ out) {
    int idx = blockIdx.x * blockDim.x + threadIdx.x;  // 0 .. B*H-1
    int b = idx >> 10, n = idx & (HID - 1);
    size_t gb = (size_t)b * H3;
    float ir = gi[gb + n], iz = gi[gb + HID + n], in_ = gi[gb + 2 * HID + n];
    float hr = gh[gb + n], hz = gh[gb + HID + n], hn = gh[gb + 2 * HID + n];
    float r = 1.0f / (1.0f + __expf(-(ir + hr)));
    float z = 1.0f / (1.0f + __expf(-(iz + hz)));
    float nn = tanhf(in_ + r * hn);
    float hp = h[idx];
    float keep = (t < length[b]) ? 1.0f : 0.0f;
    float hnew = (nn + z * (hp - nn)) * keep;
    h[idx] = hnew;
    h16[idx] = (_Float16)hnew;
    out[HS_BASE + ((size_t)t * BATCH + b) * 2048 + n] = hnew;
}

// ---------------- attention scores: one wave per (l, b) ----------------
__global__ void attn_scores(const float* __restrict__ q, const float* __restrict__ post,
                            const int* __restrict__ plen, float* __restrict__ scores) {
    int wid  = (blockIdx.x * blockDim.x + threadIdx.x) >> 5;  // 0..PLEN*B-1
    int lane = threadIdx.x & 31;
    int l = wid >> 6, b = wid & (BATCH - 1);
    const float* pv = post + ((size_t)l * BATCH + b) * PDIM;
    const float* qv = q + (size_t)b * PDIM;
    float s = 0.0f;
#pragma unroll 2
    for (int i = lane * 4; i < PDIM; i += 32 * 4) {
        float4 p4 = *(const float4*)(pv + i);
        float4 q4 = *(const float4*)(qv + i);
        s += p4.x * q4.x + p4.y * q4.y + p4.z * q4.z + p4.w * q4.w;
    }
#pragma unroll
    for (int off = 16; off > 0; off >>= 1) s += __shfl_down(s, off, 32);
    if (lane == 0) scores[l * BATCH + b] = (l < plen[b]) ? s : NEGV;
}

// ---------------- masked softmax over l (one block per batch) ----------------
__global__ void attn_softmax(const float* __restrict__ scores, float* __restrict__ attn,
                             float* __restrict__ out, int t) {
    __shared__ float red[PLEN];
    int b = blockIdx.x, l = threadIdx.x;
    float s = scores[l * BATCH + b];
    red[l] = s; __syncthreads();
#pragma unroll
    for (int st = PLEN / 2; st > 0; st >>= 1) {
        if (l < st) red[l] = fmaxf(red[l], red[l + st]);
        __syncthreads();
    }
    float m = red[0]; __syncthreads();
    float e = __expf(s - m);
    red[l] = e; __syncthreads();
#pragma unroll
    for (int st = PLEN / 2; st > 0; st >>= 1) {
        if (l < st) red[l] += red[l + st];
        __syncthreads();
    }
    float a = e / red[0];
    attn[l * BATCH + b] = a;
    out[ATTN_BASE + (size_t)t * PLEN * BATCH + (size_t)l * BATCH + b] = a;
}

// ---------------- context: hs[:, :, H:2H] ----------------
__global__ void attn_context(const float* __restrict__ attn, const float* __restrict__ post,
                             float* __restrict__ out, int t) {
    __shared__ float sa[PLEN];
    int b = blockIdx.x >> 2;
    int p = ((blockIdx.x & 3) << 8) + threadIdx.x;
    sa[threadIdx.x] = attn[threadIdx.x * BATCH + b];
    __syncthreads();
    const float* pb = post + (size_t)b * PDIM + p;
    float acc = 0.0f;
#pragma unroll 4
    for (int l = 0; l < PLEN; ++l)
        acc += sa[l] * pb[(size_t)l * BATCH * PDIM];
    out[HS_BASE + ((size_t)t * BATCH + b) * 2048 + HID + p] = acc;
}

// ---------------- host orchestration ----------------
extern "C" void kernel_launch(void* const* d_in, const int* in_sizes, int n_in,
                              void* d_out, int out_size, void* d_ws, size_t ws_size,
                              hipStream_t stream) {
    const float* incoming = (const float*)d_in[0];   // [S, B, I]
    const float* post     = (const float*)d_in[1];   // [PLEN, B, P]
    const float* h_init   = (const float*)d_in[2];   // [1, 1, H]
    const float* W_ih     = (const float*)d_in[3];   // [3H, I]
    const float* W_hh     = (const float*)d_in[4];   // [3H, H]
    const float* b_ih     = (const float*)d_in[5];   // [3H]
    const float* b_hh     = (const float*)d_in[6];   // [3H]
    const float* Wq       = (const float*)d_in[7];   // [P, H]
    const float* bq       = (const float*)d_in[8];   // [P]
    const int*   length   = (const int*)d_in[9];     // [B]
    const int*   plen     = (const int*)d_in[10];    // [B]
    float* out = (float*)d_out;

    // workspace carve-up (all 32B-aligned)
    _Float16* Wih16 = (_Float16*)d_ws;                 // 3H*I halves
    _Float16* Whh16 = Wih16 + (size_t)H3 * IN;         // 3H*H halves
    _Float16* Wq16  = Whh16 + (size_t)H3 * HID;        // P*H halves
    _Float16* x16   = Wq16  + (size_t)PDIM * HID;      // [B, I] halves
    _Float16* h16   = x16   + (size_t)BATCH * IN;      // [B, H] halves
    float* gi = (float*)(h16 + (size_t)BATCH * HID);   // [B, 3H]
    float* gh = gi + (size_t)BATCH * H3;               // [B, 3H]
    float* h  = gh + (size_t)BATCH * H3;               // [B, H]
    float* q  = h  + (size_t)BATCH * HID;              // [B, P]
    float* sc = q  + (size_t)BATCH * PDIM;             // [PLEN, B]
    float* at = sc + (size_t)PLEN * BATCH;             // [PLEN, B]

    // one-time fp16 weight conversion (weights stay L2-resident at half size)
    cvt_f32_to_f16<<<(H3 * IN  + 255) / 256, 256, 0, stream>>>(W_ih, Wih16, H3 * IN);
    cvt_f32_to_f16<<<(H3 * HID + 255) / 256, 256, 0, stream>>>(W_hh, Whh16, H3 * HID);
    cvt_f32_to_f16<<<(PDIM * HID + 255) / 256, 256, 0, stream>>>(Wq, Wq16, PDIM * HID);
    init_h<<<(BATCH * HID) / 256, 256, 0, stream>>>(h_init, h, h16);

    for (int t = 0; t < S_LEN; ++t) {
        const float* xt = incoming + (size_t)t * BATCH * IN;
        cvt_f32_to_f16<<<(BATCH * IN) / 256, 256, 0, stream>>>(xt, x16, BATCH * IN);
        // gi = x_t @ W_ih^T + b_ih ; gh = h @ W_hh^T + b_hh
        gemm_wmma_f16<<<(H3 / 128), 128, 0, stream>>>(x16, IN,  Wih16, b_ih, gi, H3, IN);
        gemm_wmma_f16<<<(H3 / 128), 128, 0, stream>>>(h16, HID, Whh16, b_hh, gh, H3, HID);
        gru_gates<<<(BATCH * HID) / 256, 256, 0, stream>>>(gi, gh, h, h16, length, t, out);
        // query = h_new @ Wq^T + bq
        gemm_wmma_f16<<<(PDIM / 128), 128, 0, stream>>>(h16, HID, Wq16, bq, q, PDIM, HID);
        attn_scores<<<(PLEN * BATCH) / 8, 256, 0, stream>>>(q, post, plen, sc);
        attn_softmax<<<BATCH, PLEN, 0, stream>>>(sc, at, out, t);
        attn_context<<<BATCH * 4, 256, 0, stream>>>(at, post, out, t);
    }
    // h_last
    hipMemcpyAsync(out, h, (size_t)BATCH * HID * sizeof(float),
                   hipMemcpyDeviceToDevice, stream);
}
